// Encoder_43568148250937
// MI455X (gfx1250) — compile-verified, matrix operand order checked by reading, not run on hardware
//
#include <hip/hip_runtime.h>

#define IN_CH 256
#define HID   128
#define OUT_CH 64

typedef __bf16 bf16x16 __attribute__((ext_vector_type(16)));
typedef float  f32x8   __attribute__((ext_vector_type(8)));

// ---------------- utility: fill ----------------
__global__ void fill_kernel(float* __restrict__ p, float v, long n) {
    long i = (long)blockIdx.x * blockDim.x + threadIdx.x;
    long stride = (long)gridDim.x * blockDim.x;
    for (; i < n; i += stride) p[i] = v;
}

// ---------------- degree (targets get +1 per in-edge; self-loop pre-seeded) ----------------
__global__ void degree_kernel(const long long* __restrict__ col, float* __restrict__ deg, int E) {
    int i = blockIdx.x * blockDim.x + threadIdx.x;
    if (i < E) atomicAdd(&deg[(int)col[i]], 1.0f);
}

__global__ void rsqrt_kernel(float* __restrict__ d, int n) {
    int i = blockIdx.x * blockDim.x + threadIdx.x;
    if (i < n) d[i] = rsqrtf(d[i]);
}

// ---------------- GEMM1: H[N,128] = X[N,256] @ W1[256,128]  (bf16 WMMA, f32 acc) ----------------
// block = 256 threads = 8 waves; block handles 16 rows x 128 cols; wave wv owns N-tile wv.
__global__ __launch_bounds__(256) void gemm1_kernel(const float* __restrict__ X,
                                                    const float* __restrict__ W,
                                                    float* __restrict__ H) {
    const int lane = threadIdx.x & 31;
    const int wv   = threadIdx.x >> 5;      // 0..7 -> 16-col tile
    const int hi   = lane >> 4;             // half-wave select
    const int lm   = lane & 15;
    const long row = (long)blockIdx.x * 16 + lm;     // A row held by this lane
    const int  ncol = wv * 16 + lm;                  // B/D column held by this lane
    f32x8 acc = {};
    const float* xa_base = X + row * IN_CH;
    for (int kk = 0; kk < IN_CH; kk += 32) {
        bf16x16 a, b;
        // A 16x32 bf16: lanes 0-15 hold K = kk+0..7 (v0-3) and kk+16..23 (v4-7);
        // lanes 16-31 hold K = kk+8..15 and kk+24..31.
        const float* xa = xa_base + kk + hi * 8;
#pragma unroll
        for (int e = 0; e < 8; ++e) { a[e] = (__bf16)xa[e]; a[8 + e] = (__bf16)xa[16 + e]; }
        // B 32x16 bf16: mirrored layout, lane holds column ncol, same K striping.
        const float* wb = W + (long)(kk + hi * 8) * HID + ncol;
#pragma unroll
        for (int e = 0; e < 8; ++e) { b[e] = (__bf16)wb[(long)e * HID]; b[8 + e] = (__bf16)wb[(long)(16 + e) * HID]; }
        acc = __builtin_amdgcn_wmma_f32_16x16x32_bf16(false, a, false, b, (short)0, acc, false, false);
    }
    // C/D: VGPR v -> row M = v + 8*hi, col = lm
    float* o = H + ((long)blockIdx.x * 16 + 8 * hi) * HID + ncol;
#pragma unroll
    for (int v = 0; v < 8; ++v) o[(long)v * HID] = acc[v];
}

// ---------------- GEMM2: H2[N,64] = H1[N,128] @ W2[128,64] ----------------
__global__ __launch_bounds__(128) void gemm2_kernel(const float* __restrict__ H1,
                                                    const float* __restrict__ W,
                                                    float* __restrict__ H2) {
    const int lane = threadIdx.x & 31;
    const int wv   = threadIdx.x >> 5;      // 0..3
    const int hi   = lane >> 4;
    const int lm   = lane & 15;
    const long row = (long)blockIdx.x * 16 + lm;
    const int  ncol = wv * 16 + lm;
    f32x8 acc = {};
    const float* ha = H1 + row * HID;
    for (int kk = 0; kk < HID; kk += 32) {
        bf16x16 a, b;
        const float* xa = ha + kk + hi * 8;
#pragma unroll
        for (int e = 0; e < 8; ++e) { a[e] = (__bf16)xa[e]; a[8 + e] = (__bf16)xa[16 + e]; }
        const float* wb = W + (long)(kk + hi * 8) * OUT_CH + ncol;
#pragma unroll
        for (int e = 0; e < 8; ++e) { b[e] = (__bf16)wb[(long)e * OUT_CH]; b[8 + e] = (__bf16)wb[(long)(16 + e) * OUT_CH]; }
        acc = __builtin_amdgcn_wmma_f32_16x16x32_bf16(false, a, false, b, (short)0, acc, false, false);
    }
    float* o = H2 + ((long)blockIdx.x * 16 + 8 * hi) * OUT_CH + ncol;
#pragma unroll
    for (int v = 0; v < 8; ++v) o[(long)v * OUT_CH] = acc[v];
}

// ---------------- edge aggregation: Acc[col] += dinv[row]*dinv[col] * Hsrc[row] ----------------
// one wave32 per edge; lane covers features lane, lane+32, ...
__global__ __launch_bounds__(256) void aggregate_kernel(const long long* __restrict__ row,
                                                        const long long* __restrict__ col,
                                                        const float* __restrict__ dinv,
                                                        const float* __restrict__ Hsrc,
                                                        float* __restrict__ Acc,
                                                        int E, int F) {
    int w = (int)(((long)blockIdx.x * blockDim.x + threadIdx.x) >> 5);
    int lane = threadIdx.x & 31;
    if (w >= E) return;
    int r = (int)row[w];
    int c = (int)col[w];
    float nrm = dinv[r] * dinv[c];
    const float* hs = Hsrc + (long)r * F;
    float* ac = Acc + (long)c * F;
    for (int f = lane; f < F; f += 32) atomicAdd(&ac[f], nrm * hs[f]);
}

// ---------------- layer1 epilogue: A = relu(A + dinv^2*H + b1) ----------------
__global__ void layer1_post_kernel(const float* __restrict__ H, const float* __restrict__ dinv,
                                   const float* __restrict__ b1, float* __restrict__ A, long n) {
    long i = (long)blockIdx.x * blockDim.x + threadIdx.x;
    long stride = (long)gridDim.x * blockDim.x;
    for (; i < n; i += stride) {
        int node = (int)(i >> 7);   // /HID
        int f    = (int)(i & (HID - 1));
        float di = dinv[node];
        float v = A[i] + di * di * H[i] + b1[f];
        A[i] = v > 0.f ? v : 0.f;
    }
}

// ---------------- layer2 epilogue: Out += dinv^2*H2 + b2 ----------------
__global__ void layer2_post_kernel(const float* __restrict__ H2, const float* __restrict__ dinv,
                                   const float* __restrict__ b2, float* __restrict__ Out, long n) {
    long i = (long)blockIdx.x * blockDim.x + threadIdx.x;
    long stride = (long)gridDim.x * blockDim.x;
    for (; i < n; i += stride) {
        int node = (int)(i >> 6);   // /OUT_CH
        int f    = (int)(i & (OUT_CH - 1));
        float di = dinv[node];
        Out[i] = Out[i] + di * di * H2[i] + b2[f];
    }
}

extern "C" void kernel_launch(void* const* d_in, const int* in_sizes, int n_in,
                              void* d_out, int out_size, void* d_ws, size_t ws_size,
                              hipStream_t stream) {
    (void)n_in; (void)out_size; (void)ws_size;
    const float*     x    = (const float*)d_in[0];
    const long long* eidx = (const long long*)d_in[1];
    const float*     W1   = (const float*)d_in[2];
    const float*     b1   = (const float*)d_in[3];
    const float*     W2   = (const float*)d_in[4];
    const float*     b2   = (const float*)d_in[5];
    float* out = (float*)d_out;

    const int N = in_sizes[0] / IN_CH;      // 100000 (divisible by 16)
    const int E = in_sizes[1] / 2;          // 1600000
    const long long* rowI = eidx;           // sources
    const long long* colI = eidx + E;       // targets

    // workspace carve-up
    char* ws = (char*)d_ws;
    size_t off = 0;
    auto carve = [&](size_t bytes) -> float* {
        off = (off + 255) & ~(size_t)255;
        float* p = (float*)(ws + off);
        off += bytes;
        return p;
    };
    float* dinv = carve((size_t)N * sizeof(float));               // deg -> dinv (in place)
    float* H    = carve((size_t)N * HID * sizeof(float));         // x @ W1
    float* A1   = carve((size_t)N * HID * sizeof(float));         // edge agg -> h1 (relu'd, in place)
    float* H2   = carve((size_t)N * OUT_CH * sizeof(float));      // h1 @ W2

    // init
    fill_kernel<<<1024, 256, 0, stream>>>(dinv, 1.0f, (long)N);                 // self-loop seed
    fill_kernel<<<4096, 256, 0, stream>>>(A1, 0.0f, (long)N * HID);
    fill_kernel<<<4096, 256, 0, stream>>>(out, 0.0f, (long)N * OUT_CH);

    // normalization
    degree_kernel<<<(E + 255) / 256, 256, 0, stream>>>(colI, dinv, E);
    rsqrt_kernel<<<(N + 255) / 256, 256, 0, stream>>>(dinv, N);

    // layer 1
    gemm1_kernel<<<N / 16, 256, 0, stream>>>(x, W1, H);
    aggregate_kernel<<<(E + 7) / 8, 256, 0, stream>>>(rowI, colI, dinv, H, A1, E, HID);
    layer1_post_kernel<<<4096, 256, 0, stream>>>(H, dinv, b1, A1, (long)N * HID);

    // layer 2
    gemm2_kernel<<<N / 16, 128, 0, stream>>>(A1, W2, H2);
    aggregate_kernel<<<(E + 7) / 8, 256, 0, stream>>>(rowI, colI, dinv, H2, out, E, OUT_CH);
    layer2_post_kernel<<<4096, 256, 0, stream>>>(H2, dinv, b2, out, (long)N * OUT_CH);
}
